// MolTransformerGCNPE_7241314861283
// MI455X (gfx1250) — compile-verified, matrix operand order checked by reading, not run on hardware
//
#include <hip/hip_runtime.h>
#include <hip/hip_bf16.h>
#include <stdint.h>

typedef __attribute__((ext_vector_type(16))) _Float16 v16h;
typedef __attribute__((ext_vector_type(8)))  float    v8f;
typedef __attribute__((ext_vector_type(4)))  int      v4i;

#define NNODES 49152
#define NEDGES 196608
#define DMODEL 256
#define BROWS  65536   // B*S
#define NHEAD  32
#define SLEN   64
#define VLEN   48

#if defined(__has_builtin)
#if __has_builtin(__builtin_amdgcn_global_load_async_to_lds_b128)
#define HAVE_ASYNC_LDS 1
#endif
#endif

#if HAVE_ASYNC_LDS
typedef __attribute__((address_space(1))) v4i gas_v4i;  // global int4
typedef __attribute__((address_space(3))) v4i las_v4i;  // LDS int4
#endif

// ---------------------------------------------------------------------------
// Utility kernels
// ---------------------------------------------------------------------------
__global__ void fill_u32_kernel(uint32_t* __restrict__ p, size_t n, uint32_t v) {
  size_t i = (size_t)blockIdx.x * blockDim.x + threadIdx.x;
  if (i < n) p[i] = v;
}

__global__ void cvt_f32_to_f16_kernel(const float* __restrict__ in,
                                      _Float16* __restrict__ out, size_t n) {
  size_t i = (size_t)blockIdx.x * blockDim.x + threadIdx.x;
  if (i < n) out[i] = (_Float16)in[i];
}

// Transpose K x Nc (row-major) f32 weights into Nc x K (row-major) f16 so B
// fragments become two contiguous 16B loads per lane (no LDS scatter needed).
__global__ void cvt_transpose_kernel(const float* __restrict__ in,
                                     _Float16* __restrict__ out, int K, int Nc) {
  int idx = blockIdx.x * 256 + threadIdx.x;
  if (idx < K * Nc) {
    int k = idx / Nc, n = idx - k * Nc;
    out[(size_t)n * K + k] = (_Float16)in[idx];
  }
}

// ---------------------------------------------------------------------------
// Embedding gather + norm clip:  e = emb[atom] * min(1, 1/(||.||+1e-7))
// ---------------------------------------------------------------------------
__global__ __launch_bounds__(256) void embed_norm_kernel(
    const int* __restrict__ atoms, const float* __restrict__ emb,
    float* __restrict__ e) {
  const int node = blockIdx.x * 8 + (threadIdx.x >> 5);
  const int lane = threadIdx.x & 31;
  const float* row = emb + (size_t)atoms[node] * DMODEL + lane * 8;
  float x[8];
  float ss = 0.f;
#pragma unroll
  for (int j = 0; j < 8; ++j) { x[j] = row[j]; ss += x[j] * x[j]; }
#pragma unroll
  for (int off = 16; off > 0; off >>= 1) ss += __shfl_xor(ss, off, 32);
  const float sc = fminf(1.f, 1.f / (sqrtf(ss) + 1e-7f));
  float* dst = e + (size_t)node * DMODEL + lane * 8;
#pragma unroll
  for (int j = 0; j < 8; ++j) dst[j] = x[j] * sc;
}

// ---------------------------------------------------------------------------
// Graph prep
// ---------------------------------------------------------------------------
__global__ void deg_kernel(const int* __restrict__ dstI, float* __restrict__ deg) {
  int i = blockIdx.x * 256 + threadIdx.x;
  if (i < NEDGES) atomicAdd(&deg[dstI[i]], 1.f);
}
__global__ void dinv_kernel(const float* __restrict__ deg, float* __restrict__ dinv) {
  int i = blockIdx.x * 256 + threadIdx.x;
  if (i < NNODES) dinv[i] = rsqrtf(deg[i] + 1.f);
}
__global__ void csum_kernel(const int* __restrict__ srcI, const int* __restrict__ dstI,
                            const float* __restrict__ dinv, float* __restrict__ csum) {
  int i = blockIdx.x * 256 + threadIdx.x;
  if (i < NEDGES) atomicAdd(&csum[dstI[i]], dinv[srcI[i]]);
}

// pe1[n,d] = relu(conv1_w[d] * dinv[n]*(csum[n]+dinv[n]) + conv1_b[d])
__global__ void gcn1_kernel(const float* __restrict__ dinv, const float* __restrict__ csum,
                            const float* __restrict__ w1, const float* __restrict__ b1,
                            _Float16* __restrict__ pe1) {
  size_t idx = (size_t)blockIdx.x * 256 + threadIdx.x;
  int n = (int)(idx >> 8), d = (int)(idx & 255);
  float a = dinv[n] * (csum[n] + dinv[n]);
  pe1[idx] = (_Float16)fmaxf(w1[d] * a + b1[d], 0.f);
}

__global__ void scatter_kernel(const int* __restrict__ srcI, const int* __restrict__ dstI,
                               const float* __restrict__ dinv, const float* __restrict__ hmid,
                               float* __restrict__ agg) {
  size_t idx = (size_t)blockIdx.x * 256 + threadIdx.x;  // E*256 threads
  int e = (int)(idx >> 8), c = (int)(idx & 255);
  int sN = srcI[e], dN = dstI[e];
  atomicAdd(&agg[(size_t)dN * DMODEL + c],
            hmid[(size_t)sN * DMODEL + c] * (dinv[sN] * dinv[dN]));
}

__global__ void combine_kernel(const float* __restrict__ e, const float* __restrict__ agg,
                               const float* __restrict__ hmid, const float* __restrict__ dinv,
                               const float* __restrict__ b2, _Float16* __restrict__ xp) {
  size_t idx = (size_t)blockIdx.x * 256 + threadIdx.x;  // N*256
  int n = (int)(idx >> 8), d = (int)(idx & 255);
  float di = dinv[n];
  float val = e[idx] + agg[idx] + hmid[idx] * di * di + b2[d];
  int bb = n / VLEN, s = n - bb * VLEN;
  xp[((size_t)(bb * SLEN + s)) * DMODEL + d] = (_Float16)val;
}

// ---------------------------------------------------------------------------
// WMMA GEMM:  out = A(MxK,f16) @ Wt^T  where Wt is Nc x K (pre-transposed f16)
// workgroup tile 128x128, 8 waves, 32x64 per wave = 2x4 v_wmma_f32_16x16x32_f16
// A staged via async-to-LDS (gfx1250) with s_wait_asynccnt; B fragments straight
// from global (weights are L2/WGP$ resident).
// ---------------------------------------------------------------------------
__global__ __launch_bounds__(256) void gemm_wmma_kernel(
    const _Float16* __restrict__ A, const _Float16* __restrict__ Wt,
    const float* __restrict__ bias, const float* __restrict__ resid,
    float* __restrict__ outF, _Float16* __restrict__ outH,
    int M, int K, int Nc, int doRelu) {
  __shared__ alignas(16) _Float16 sa[128 * 32];

  const int tid = threadIdx.x, lane = tid & 31, wid = tid >> 5;
  const int wm = wid & 3;        // 4 M-groups of 32 rows
  const int wn = wid >> 2;       // 2 N-groups of 64 cols
  const int m0 = blockIdx.x * 128, n0 = blockIdx.y * 128;

  const int arow = tid >> 1;         // 0..127
  const int akp  = (tid & 1) << 4;   // 0 or 16 halves

  v8f z = {0.f, 0.f, 0.f, 0.f, 0.f, 0.f, 0.f, 0.f};
  v8f acc[2][4];
#pragma unroll
  for (int i = 0; i < 2; ++i)
#pragma unroll
    for (int j = 0; j < 4; ++j) acc[i][j] = z;

  const int rbase = (lane >> 4) << 3;  // ISA fragment K/row group
  const int lm = lane & 15;

  for (int k0 = 0; k0 < K; k0 += 32) {
    // ---- stage A tile 128x32 into LDS ----
    const _Float16* gsrc = A + (size_t)(m0 + arow) * K + k0 + akp;
#if HAVE_ASYNC_LDS
    __builtin_amdgcn_global_load_async_to_lds_b128(
        (gas_v4i*)(void*)gsrc, (las_v4i*)(void*)&sa[arow * 32 + akp], 0, 0);
    __builtin_amdgcn_global_load_async_to_lds_b128(
        (gas_v4i*)(void*)(gsrc + 8), (las_v4i*)(void*)&sa[arow * 32 + akp + 8], 0, 0);
    asm volatile("s_wait_asynccnt 0x0" ::: "memory");
#else
    {
      const uint4* ga = reinterpret_cast<const uint4*>(gsrc);
      uint4 a0 = ga[0], a1 = ga[1];
      *reinterpret_cast<uint4*>(&sa[arow * 32 + akp])     = a0;
      *reinterpret_cast<uint4*>(&sa[arow * 32 + akp + 8]) = a1;
    }
#endif
    if (k0 + 32 < K)  // gfx1250 global_prefetch_b8 for next A tile
      __builtin_prefetch(gsrc + 32, 0, 0);
    __syncthreads();

    // ---- fragments ----
    union Frag { uint4 u[2]; v16h v; };
    Frag af[2], bf[4];
#pragma unroll
    for (int t = 0; t < 2; ++t) {
      const _Float16* pa = &sa[(wm * 32 + t * 16 + lm) * 32 + rbase];
      af[t].u[0] = *reinterpret_cast<const uint4*>(pa);
      af[t].u[1] = *reinterpret_cast<const uint4*>(pa + 16);
    }
#pragma unroll
    for (int t = 0; t < 4; ++t) {
      const _Float16* pb = Wt + (size_t)(n0 + wn * 64 + t * 16 + lm) * K + k0 + rbase;
      bf[t].u[0] = *reinterpret_cast<const uint4*>(pb);
      bf[t].u[1] = *reinterpret_cast<const uint4*>(pb + 16);
    }
#pragma unroll
    for (int tm = 0; tm < 2; ++tm)
#pragma unroll
      for (int tn = 0; tn < 4; ++tn)
        acc[tm][tn] = __builtin_amdgcn_wmma_f32_16x16x32_f16(
            false, af[tm].v, false, bf[tn].v, (short)0, acc[tm][tn], false, false);
    __syncthreads();
  }

  // Epilogue: C layout — lane col = lm, VGPR r -> row r + 8*(lane>=16)
#pragma unroll
  for (int tm = 0; tm < 2; ++tm) {
#pragma unroll
    for (int tn = 0; tn < 4; ++tn) {
      const int n = n0 + wn * 64 + tn * 16 + lm;
      const float bv = bias ? bias[n] : 0.f;
#pragma unroll
      for (int r = 0; r < 8; ++r) {
        const int m = m0 + wm * 32 + tm * 16 + rbase + r;
        float v = acc[tm][tn][r] + bv;
        if (resid) v += resid[(size_t)m * Nc + n];
        if (doRelu) v = fmaxf(v, 0.f);
        if (outF) outF[(size_t)m * Nc + n] = v;
        if (outH) outH[(size_t)m * Nc + n] = (_Float16)v;
      }
    }
  }
}

// ---------------------------------------------------------------------------
// Attention: block per (b, head), 64 threads (one per query). HD=8 -> VALU.
// Mask applies to QUERIES only (reference broadcast semantics); keys unmasked.
// ---------------------------------------------------------------------------
__global__ __launch_bounds__(64) void attn_kernel(const _Float16* __restrict__ qkv,
                                                  _Float16* __restrict__ obuf) {
  const int b = blockIdx.x >> 5;
  const int head = blockIdx.x & 31;
  const int s = threadIdx.x;
  __shared__ float ks[SLEN][8];
  __shared__ float vs[SLEN][8];
  const size_t base = (size_t)(b * SLEN + s) * (3 * DMODEL) + head * 8;
  float q[8];
#pragma unroll
  for (int d = 0; d < 8; ++d) {
    q[d]     = (float)qkv[base + d];
    ks[s][d] = (float)qkv[base + DMODEL + d];
    vs[s][d] = (float)qkv[base + 2 * DMODEL + d];
  }
  __syncthreads();
  const float scale = 0.3535533905932738f;  // 1/sqrt(8)
  const bool valid = s < VLEN;
  float sc[SLEN];
  float mx = -3.4e38f;
#pragma unroll 4
  for (int k = 0; k < SLEN; ++k) {
    float d0 = 0.f;
#pragma unroll
    for (int d = 0; d < 8; ++d) d0 += q[d] * ks[k][d];
    float v = valid ? d0 * scale : -1e9f;
    sc[k] = v;
    mx = fmaxf(mx, v);
  }
  float sum = 0.f;
#pragma unroll 4
  for (int k = 0; k < SLEN; ++k) { sc[k] = __expf(sc[k] - mx); sum += sc[k]; }
  const float inv = 1.f / sum;
  float o[8] = {0, 0, 0, 0, 0, 0, 0, 0};
#pragma unroll 4
  for (int k = 0; k < SLEN; ++k) {
    const float p = sc[k] * inv;
#pragma unroll
    for (int d = 0; d < 8; ++d) o[d] += p * vs[k][d];
  }
  const size_t ob = (size_t)(b * SLEN + s) * DMODEL + head * 8;
#pragma unroll
  for (int d = 0; d < 8; ++d) obuf[ob + d] = (_Float16)o[d];
}

// ---------------------------------------------------------------------------
// LayerNorm: one wave per row of 256, shfl_xor reduction (wave32)
// ---------------------------------------------------------------------------
__global__ __launch_bounds__(32) void ln_kernel(
    const float* __restrict__ src, const float* __restrict__ gamma,
    const float* __restrict__ beta, float* __restrict__ outF,
    _Float16* __restrict__ outH) {
  const int row = blockIdx.x, lane = threadIdx.x;
  const size_t base = (size_t)row * DMODEL + lane * 8;
  float x[8];
  float s = 0.f, ss = 0.f;
#pragma unroll
  for (int j = 0; j < 8; ++j) { x[j] = src[base + j]; s += x[j]; ss += x[j] * x[j]; }
#pragma unroll
  for (int off = 16; off > 0; off >>= 1) {
    s  += __shfl_xor(s, off, 32);
    ss += __shfl_xor(ss, off, 32);
  }
  const float mu  = s * (1.f / 256.f);
  const float var = ss * (1.f / 256.f) - mu * mu;
  const float inv = rsqrtf(var + 1e-5f);
#pragma unroll
  for (int j = 0; j < 8; ++j) {
    const int c = lane * 8 + j;
    const float y = (x[j] - mu) * inv * gamma[c] + beta[c];
    outF[base + j] = y;
    if (outH) outH[base + j] = (_Float16)y;
  }
}

// ---------------------------------------------------------------------------
// Host orchestration
// ---------------------------------------------------------------------------
extern "C" void kernel_launch(void* const* d_in, const int* in_sizes, int n_in,
                              void* d_out, int out_size, void* d_ws, size_t ws_size,
                              hipStream_t stream) {
  (void)in_sizes; (void)n_in; (void)out_size; (void)ws_size;
  const int*   x_atoms = (const int*)d_in[0];
  const int*   edges   = (const int*)d_in[1];
  const int*   srcI    = edges;
  const int*   dstI    = edges + NEDGES;
  const float* emb     = (const float*)d_in[3];
  const float* conv1_w = (const float*)d_in[4];
  const float* conv1_b = (const float*)d_in[5];
  const float* conv2_w = (const float*)d_in[6];
  const float* conv2_b = (const float*)d_in[7];
  const float* in_w    = (const float*)d_in[8];
  const float* in_b    = (const float*)d_in[9];
  const float* attn_w  = (const float*)d_in[10];
  const float* attn_b  = (const float*)d_in[11];
  const float* proj_w  = (const float*)d_in[12];
  const float* proj_b  = (const float*)d_in[13];
  const float* ln1_s   = (const float*)d_in[14];
  const float* ln1_b   = (const float*)d_in[15];
  const float* ff1_w   = (const float*)d_in[16];
  const float* ff1_b   = (const float*)d_in[17];
  const float* ff2_w   = (const float*)d_in[18];
  const float* ff2_b   = (const float*)d_in[19];
  const float* ln2_s   = (const float*)d_in[20];
  const float* ln2_b   = (const float*)d_in[21];

  char* wsb = (char*)d_ws;
  size_t off = 0;
  auto alloc = [&](size_t bytes) -> void* {
    void* p = wsb + off;
    off += (bytes + 255) & ~(size_t)255;
    return p;
  };
  float*    e_buf  = (float*)alloc((size_t)NNODES * DMODEL * 4);
  float*    deg    = (float*)alloc((size_t)NNODES * 4);
  float*    dinv   = (float*)alloc((size_t)NNODES * 4);
  float*    csum   = (float*)alloc((size_t)NNODES * 4);
  _Float16* pe1    = (_Float16*)alloc((size_t)NNODES * DMODEL * 2);
  float*    hmid   = (float*)alloc((size_t)NNODES * DMODEL * 4);
  float*    agg    = (float*)alloc((size_t)NNODES * DMODEL * 4);
  _Float16* xp     = (_Float16*)alloc((size_t)BROWS * DMODEL * 2);
  float*    hbuf   = (float*)alloc((size_t)BROWS * DMODEL * 4);
  _Float16* hh     = (_Float16*)alloc((size_t)BROWS * DMODEL * 2);
  _Float16* qkv    = (_Float16*)alloc((size_t)BROWS * 3 * DMODEL * 2);
  _Float16* obuf   = (_Float16*)alloc((size_t)BROWS * DMODEL * 2);
  _Float16* ubuf   = (_Float16*)alloc((size_t)BROWS * DMODEL * 2);
  float*    tmp    = (float*)alloc((size_t)BROWS * DMODEL * 4);
  _Float16* conv2h = (_Float16*)alloc((size_t)DMODEL * DMODEL * 2);
  _Float16* in_wh  = (_Float16*)alloc((size_t)DMODEL * DMODEL * 2);
  _Float16* attn_wh= (_Float16*)alloc((size_t)3 * DMODEL * 3 * DMODEL * 2);
  _Float16* proj_wh= (_Float16*)alloc((size_t)3 * DMODEL * DMODEL * 2);
  _Float16* ff1_wh = (_Float16*)alloc((size_t)3 * DMODEL * DMODEL * 2);
  _Float16* ff2_wh = (_Float16*)alloc((size_t)3 * DMODEL * DMODEL * 2);

  const size_t WDD  = (size_t)DMODEL * DMODEL;        // 65536
  const size_t WQKV = (size_t)DMODEL * 3 * DMODEL;    // 196608
  const int    TB_DD  = (int)((WDD + 255) / 256);
  const int    TB_QKV = (int)((WQKV + 255) / 256);

  // Weight f32 -> transposed f16 (Nc x K layout for direct B-fragment loads)
  cvt_transpose_kernel<<<TB_DD, 256, 0, stream>>>(conv2_w, conv2h, DMODEL, DMODEL);
  cvt_transpose_kernel<<<TB_DD, 256, 0, stream>>>(in_w, in_wh, DMODEL, DMODEL);
  for (int l = 0; l < 3; ++l) {
    cvt_transpose_kernel<<<TB_QKV, 256, 0, stream>>>(
        attn_w + (size_t)l * WQKV, attn_wh + (size_t)l * WQKV, DMODEL, 3 * DMODEL);
    cvt_transpose_kernel<<<TB_DD, 256, 0, stream>>>(
        proj_w + (size_t)l * WDD, proj_wh + (size_t)l * WDD, DMODEL, DMODEL);
    cvt_transpose_kernel<<<TB_DD, 256, 0, stream>>>(
        ff1_w + (size_t)l * WDD, ff1_wh + (size_t)l * WDD, DMODEL, DMODEL);
    cvt_transpose_kernel<<<TB_DD, 256, 0, stream>>>(
        ff2_w + (size_t)l * WDD, ff2_wh + (size_t)l * WDD, DMODEL, DMODEL);
  }

  // Zero init (deg, csum, agg, padded x)
  fill_u32_kernel<<<(NNODES + 255) / 256, 256, 0, stream>>>((uint32_t*)deg, NNODES, 0u);
  fill_u32_kernel<<<(NNODES + 255) / 256, 256, 0, stream>>>((uint32_t*)csum, NNODES, 0u);
  fill_u32_kernel<<<(int)(((size_t)NNODES * DMODEL + 255) / 256), 256, 0, stream>>>(
      (uint32_t*)agg, (size_t)NNODES * DMODEL, 0u);
  fill_u32_kernel<<<(int)(((size_t)BROWS * DMODEL / 2 + 255) / 256), 256, 0, stream>>>(
      (uint32_t*)xp, (size_t)BROWS * DMODEL / 2, 0u);

  // Embedding + norm clip
  embed_norm_kernel<<<NNODES / 8, 256, 0, stream>>>(x_atoms, emb, e_buf);

  // Graph prep
  deg_kernel<<<(NEDGES + 255) / 256, 256, 0, stream>>>(dstI, deg);
  dinv_kernel<<<(NNODES + 255) / 256, 256, 0, stream>>>(deg, dinv);
  csum_kernel<<<(NEDGES + 255) / 256, 256, 0, stream>>>(srcI, dstI, dinv, csum);

  // GCN layer 1 (closed form for ones-feature)
  gcn1_kernel<<<(int)((size_t)NNODES * DMODEL / 256), 256, 0, stream>>>(
      dinv, csum, conv1_w, conv1_b, pe1);

  // GCN layer 2: hmid = pe1 @ conv2_w  (WMMA)
  gemm_wmma_kernel<<<dim3(NNODES / 128, DMODEL / 128), 256, 0, stream>>>(
      pe1, conv2h, nullptr, nullptr, hmid, nullptr, NNODES, DMODEL, DMODEL, 0);
  scatter_kernel<<<(int)((size_t)NEDGES * DMODEL / 256), 256, 0, stream>>>(
      srcI, dstI, dinv, hmid, agg);
  combine_kernel<<<(int)((size_t)NNODES * DMODEL / 256), 256, 0, stream>>>(
      e_buf, agg, hmid, dinv, conv2_b, xp);

  // Input projection: h = xp @ in_w + in_b   (f32 residual + f16 copy)
  gemm_wmma_kernel<<<dim3(BROWS / 128, DMODEL / 128), 256, 0, stream>>>(
      xp, in_wh, in_b, nullptr, hbuf, hh, BROWS, DMODEL, DMODEL, 0);

  // Transformer layers
  for (int l = 0; l < 3; ++l) {
    gemm_wmma_kernel<<<dim3(BROWS / 128, (3 * DMODEL) / 128), 256, 0, stream>>>(
        hh, attn_wh + (size_t)l * WQKV, attn_b + (size_t)l * 3 * DMODEL, nullptr,
        nullptr, qkv, BROWS, DMODEL, 3 * DMODEL, 0);

    attn_kernel<<<1024 * NHEAD, 64, 0, stream>>>(qkv, obuf);

    gemm_wmma_kernel<<<dim3(BROWS / 128, DMODEL / 128), 256, 0, stream>>>(
        obuf, proj_wh + (size_t)l * WDD, proj_b + (size_t)l * DMODEL, hbuf,
        tmp, nullptr, BROWS, DMODEL, DMODEL, 0);
    ln_kernel<<<BROWS, 32, 0, stream>>>(tmp, ln1_s + (size_t)l * DMODEL,
                                        ln1_b + (size_t)l * DMODEL, hbuf, hh);

    gemm_wmma_kernel<<<dim3(BROWS / 128, DMODEL / 128), 256, 0, stream>>>(
        hh, ff1_wh + (size_t)l * WDD, ff1_b + (size_t)l * DMODEL, nullptr,
        nullptr, ubuf, BROWS, DMODEL, DMODEL, 1);
    gemm_wmma_kernel<<<dim3(BROWS / 128, DMODEL / 128), 256, 0, stream>>>(
        ubuf, ff2_wh + (size_t)l * WDD, ff2_b + (size_t)l * DMODEL, hbuf,
        tmp, nullptr, BROWS, DMODEL, DMODEL, 0);
    float* lnOut = (l == 2) ? (float*)d_out : hbuf;
    ln_kernel<<<BROWS, 32, 0, stream>>>(tmp, ln2_s + (size_t)l * DMODEL,
                                        ln2_b + (size_t)l * DMODEL, lnOut, hh);
  }
}
